// CloMSFM_35132832481638
// MI455X (gfx1250) — matrix operand order repaired and checked by reading, not compiled, not run on hardware
//
#include <hip/hip_runtime.h>
#include <math.h>

#define HW      1024
#define CIN     256
#define QKV_CH  384
#define HF      128
#define NB      16
#define SCALOR  0.17677669529663687f

typedef __attribute__((ext_vector_type(16))) _Float16 v16h;
typedef __attribute__((ext_vector_type(8)))  float    v8f;

__device__ __forceinline__ v8f wmma16x16x32(v16h a, v16h b, v8f c) {
  return __builtin_amdgcn_wmma_f32_16x16x32_f16(false, a, false, b, (short)0, c,
                                                false, false);
}

// Pre-swizzled f16 A operand: [tile][kstep][lane][16] -> one aligned 32B load.
__device__ __forceinline__ v16h load_a_swz(const _Float16* __restrict__ Wz,
                                           int tile, int ksteps, int ks, int lane) {
  return *reinterpret_cast<const v16h*>(
      Wz + ((size_t)(tile * ksteps + ks) * 32 + lane) * 16);
}

// B fragment (32x16 f16, K x N) from LDS panel[n][k] (stride ldk f16 elems).
__device__ __forceinline__ v16h load_b_lds(const _Float16* __restrict__ P,
                                           int ldk, int k0, int lane) {
  const _Float16* col = P + (size_t)(lane & 15) * ldk + k0 + ((lane & 16) ? 8 : 0);
  v16h b;
#pragma unroll
  for (int i = 0; i < 8; ++i) {
    int k = 2 * i + ((i >= 4) ? 8 : 0);
    b[2 * i]     = col[k];
    b[2 * i + 1] = col[k + 1];
  }
  return b;
}

// ---------------------------------------------------------------------------
// Weight pre-swizzle: f32 row-major -> f16 A-fragment order.
// ---------------------------------------------------------------------------
__global__ __launch_bounds__(32) void k_prep_w(const float* __restrict__ W, int ld,
                                               int ksteps, _Float16* __restrict__ dst) {
  const int tile = blockIdx.x, ks = blockIdx.y, lane = threadIdx.x;
  const float* row = W + (size_t)(tile * 16 + (lane & 15)) * ld + ks * 32;
  const int khi = (lane & 16) ? 8 : 0;
  _Float16* d = dst + ((size_t)(tile * ksteps + ks) * 32 + lane) * 16;
#pragma unroll
  for (int i = 0; i < 8; ++i) {
    int k = 2 * i + ((i >= 4) ? 8 : 0) + khi;
    d[2 * i]     = (_Float16)row[k];
    d[2 * i + 1] = (_Float16)row[k + 1];
  }
}

// Fused input weights (768 rows: qkv | gq | gkv) into one swizzled buffer.
__global__ __launch_bounds__(32) void k_prep_fused_w(
    const float* __restrict__ qkv_w, const float* __restrict__ gq_w,
    const float* __restrict__ gkv_w, _Float16* __restrict__ dst) {
  const int tile = blockIdx.x, ks = blockIdx.y, lane = threadIdx.x;
  const int mrow = tile * 16;
  const float* W;
  if (mrow < QKV_CH)            W = qkv_w + (size_t)mrow * CIN;
  else if (mrow < QKV_CH + HF)  W = gq_w  + (size_t)(mrow - QKV_CH) * CIN;
  else                          W = gkv_w + (size_t)(mrow - QKV_CH - HF) * CIN;
  const float* row = W + (size_t)(lane & 15) * CIN + ks * 32;
  const int khi = (lane & 16) ? 8 : 0;
  _Float16* d = dst + ((size_t)(tile * 8 + ks) * 32 + lane) * 16;
#pragma unroll
  for (int i = 0; i < 8; ++i) {
    int k = 2 * i + ((i >= 4) ? 8 : 0) + khi;
    d[2 * i]     = (_Float16)row[k];
    d[2 * i + 1] = (_Float16)row[k + 1];
  }
}

// ---------------------------------------------------------------------------
// K1: fused input projections, M=768 fused rows, K=256, N=16/block.
// ---------------------------------------------------------------------------
__global__ __launch_bounds__(256) void k_fused_in_gemm(
    const float* __restrict__ x, const _Float16* __restrict__ wz,
    const float* __restrict__ qkv_b, const float* __restrict__ gq_b,
    const float* __restrict__ gkv_b,
    float* __restrict__ qkv_out, float* __restrict__ g_out) {
  __shared__ _Float16 pan[16][258];   // [n][k], 129-dword stride: conflict-free
  const int hw0  = blockIdx.x * 16;
  const int mblk = blockIdx.y;        // 0..5
  const int b    = blockIdx.z;
  const float* xb = x + (size_t)b * CIN * HW;

  {
    int n  = threadIdx.x & 15;
    int kb = (threadIdx.x >> 4) * 16;
#pragma unroll
    for (int i = 0; i < 16; ++i)
      pan[n][kb + i] = (_Float16)xb[(size_t)(kb + i) * HW + hw0 + n];
  }
  __syncthreads();

  const int wave = threadIdx.x >> 5;
  const int lane = threadIdx.x & 31;
  const int mrow = mblk * 128 + wave * 16;   // fused row base (wave-uniform)
  const int tile = mrow >> 4;

  const float* bias; float* outp; int orow;
  if (mrow < QKV_CH)           { bias = qkv_b + mrow;               outp = qkv_out; orow = mrow; }
  else if (mrow < QKV_CH + HF) { bias = gq_b + (mrow - QKV_CH);     outp = g_out;   orow = mrow - QKV_CH; }
  else                         { bias = gkv_b + (mrow - QKV_CH - HF); outp = g_out; orow = mrow - QKV_CH; }
  __builtin_prefetch(wz + (size_t)tile * 8 * 32 * 16, 0, 3);   // global_prefetch_b8

  v8f acc = {};
#pragma unroll
  for (int ks = 0; ks < 8; ++ks) {
    v16h a  = load_a_swz(wz, tile, 8, ks, lane);
    v16h bb = load_b_lds(&pan[0][0], 258, ks * 32, lane);
    acc = wmma16x16x32(a, bb, acc);
  }

  float* ob = outp + (size_t)b * QKV_CH * HW;
  const int n  = lane & 15;
  const int mh = (lane & 16) ? 8 : 0;
#pragma unroll
  for (int r = 0; r < 8; ++r) {
    int m = r + mh;
    ob[(size_t)(orow + m) * HW + hw0 + n] = acc[r] + bias[m];
  }
}

// ---------------------------------------------------------------------------
// K2: depthwise 3x3 'SAME' conv over 384 channels.
// ---------------------------------------------------------------------------
__global__ __launch_bounds__(256) void k_dwconv(
    const float* __restrict__ in, const float* __restrict__ w,
    const float* __restrict__ bias, float* __restrict__ out) {
  int t  = blockIdx.x * 256 + threadIdx.x;
  int hw = t & (HW - 1);
  int bc = t >> 10;
  int ch = bc % QKV_CH;
  int yy = hw >> 5, xx = hw & 31;
  const float* p  = in + (size_t)bc * HW;
  const float* wc = w + ch * 9;
  float s = bias[ch];
#pragma unroll
  for (int ky = 0; ky < 3; ++ky) {
    int y = yy + ky - 1;
    if ((unsigned)y >= 32u) continue;
#pragma unroll
    for (int kx = 0; kx < 3; ++kx) {
      int x = xx + kx - 1;
      if ((unsigned)x >= 32u) continue;
      s += wc[ky * 3 + kx] * p[y * 32 + x];
    }
  }
  out[t] = s;
}

// ---------------------------------------------------------------------------
// K3: high-freq branch: q*k -> GEMM(w1)+SiLU -> GEMM(w2) -> tanh(.)*v.
// ---------------------------------------------------------------------------
__global__ __launch_bounds__(256) void k_highfreq(
    const float* __restrict__ qkv,
    const _Float16* __restrict__ w1z, const float* __restrict__ b1,
    const _Float16* __restrict__ w2z, const float* __restrict__ b2,
    float* __restrict__ comb) {
  __shared__ _Float16 panT[16][130];
  __shared__ _Float16 panA[16][130];
  const int hw0 = blockIdx.x * 16;
  const int b   = blockIdx.y;
  const float* qb = qkv + (size_t)b * QKV_CH * HW;

  {
    int n  = threadIdx.x & 15;
    int kb = (threadIdx.x >> 4) * 8;
#pragma unroll
    for (int i = 0; i < 8; ++i) {
      int c = kb + i;
      float qv = qb[(size_t)c * HW + hw0 + n];
      float kv = qb[(size_t)(HF + c) * HW + hw0 + n];
      panT[n][c] = (_Float16)(qv * kv);
    }
  }
  __syncthreads();

  const int wave = threadIdx.x >> 5;
  const int lane = threadIdx.x & 31;
  const int m0   = wave * 16;
  const int n    = lane & 15;
  const int mh   = (lane & 16) ? 8 : 0;

  v8f acc = {};
#pragma unroll
  for (int ks = 0; ks < 4; ++ks) {
    v16h a  = load_a_swz(w1z, wave, 4, ks, lane);
    v16h bb = load_b_lds(&panT[0][0], 130, ks * 32, lane);
    acc = wmma16x16x32(a, bb, acc);
  }
#pragma unroll
  for (int r = 0; r < 8; ++r) {
    int m = m0 + r + mh;
    float v = acc[r] + b1[m];
    panA[n][m] = (_Float16)(v / (1.0f + __expf(-v)));
  }
  __syncthreads();

  v8f acc2 = {};
#pragma unroll
  for (int ks = 0; ks < 4; ++ks) {
    v16h a  = load_a_swz(w2z, wave, 4, ks, lane);
    v16h bb = load_b_lds(&panA[0][0], 130, ks * 32, lane);
    acc2 = wmma16x16x32(a, bb, acc2);
  }
  float* cb = comb + (size_t)b * CIN * HW;
#pragma unroll
  for (int r = 0; r < 8; ++r) {
    int m = m0 + r + mh;
    float a2 = acc2[r] + b2[m];
    float vv = qb[(size_t)(2 * HF + m) * HW + hw0 + n];
    cb[(size_t)m * HW + hw0 + n] = tanhf(a2 * SCALOR) * vv;
  }
}

// ---------------------------------------------------------------------------
// K4: flash attention; K/V chunks staged into LDS with async global->LDS DMA.
// kvpan rows [0,32) = K d-rows, [32,64) = V d-rows; 32 keys per chunk.
// ---------------------------------------------------------------------------
__global__ __launch_bounds__(256) void k_lowfreq_attn(
    const float* __restrict__ g, float* __restrict__ comb) {
  __shared__ float    kvpan[64][34];     // f32 panel, stride 34 dwords
  __shared__ _Float16 pbuf[8][16][34];
  const int qb = blockIdx.x;             // 0..7
  const int h  = blockIdx.y;             // 0..3
  const int b  = blockIdx.z;
  const float* gq = g + (size_t)b * QKV_CH * HW + (size_t)(h * 32) * HW;
  const float* kk = gq + (size_t)HF * HW;
  const float* vv = gq + (size_t)(2 * HF) * HW;

  const int wave  = threadIdx.x >> 5;
  const int lane  = threadIdx.x & 31;
  const int q0    = qb * 128 + wave * 16;
  const int n     = lane & 15;
  const int khi8  = (lane & 16) ? 8 : 0;
  const int khi16 = (lane & 16) ? 16 : 0;

  // Q A-fragment (scale folded), loaded once.
  v16h qa;
#pragma unroll
  for (int i = 0; i < 8; ++i) {
    int k = 2 * i + ((i >= 4) ? 8 : 0) + khi8;
    qa[2 * i]     = (_Float16)(gq[(size_t)k * HW + q0 + n] * SCALOR);
    qa[2 * i + 1] = (_Float16)(gq[(size_t)(k + 1) * HW + q0 + n] * SCALOR);
  }

  // Per-thread async-copy assignment: 2 x B128 (4 floats) segments.
  const unsigned ldsbase = (unsigned)(size_t)&kvpan[0][0];
  const int seg0 = threadIdx.x * 2;

  float mx[8], ls[8];
  v8f o0 = {}, o1 = {};
#pragma unroll
  for (int r = 0; r < 8; ++r) { mx[r] = -1e30f; ls[r] = 0.0f; }

  for (int kc = 0; kc < HW; kc += 32) {
    // ---- async stage K/V chunk into LDS (16B per lane per op) ----
#pragma unroll
    for (int s = 0; s < 2; ++s) {
      int seg = seg0 + s;
      int row = seg >> 3;                // 0..63
      int k4  = (seg & 7) * 4;           // key offset 0..28
      const float* gp = (row < 32 ? kk + (size_t)row * HW
                                  : vv + (size_t)(row - 32) * HW) + kc + k4;
      unsigned lo = ldsbase + (unsigned)(row * 34 + k4) * 4u;
      asm volatile("global_load_async_to_lds_b128 %0, %1, off"
                   :: "v"(lo), "v"((unsigned long long)(size_t)gp) : "memory");
    }
    asm volatile("s_wait_asynccnt 0x0" ::: "memory");
    __syncthreads();                      // kvpan ready for all waves

    // K^T B-fragments (k = dh rows, col = key)
    v16h kb0, kb1;
#pragma unroll
    for (int i = 0; i < 8; ++i) {
      int d = 2 * i + ((i >= 4) ? 8 : 0) + khi8;
      kb0[2 * i]     = (_Float16)kvpan[d][n];
      kb0[2 * i + 1] = (_Float16)kvpan[d + 1][n];
      kb1[2 * i]     = (_Float16)kvpan[d][16 + n];
      kb1[2 * i + 1] = (_Float16)kvpan[d + 1][16 + n];
    }
    v8f z = {};
    v8f s0 = wmma16x16x32(qa, kb0, z);
    v8f s1 = wmma16x16x32(qa, kb1, z);

#pragma unroll
    for (int r = 0; r < 8; ++r) {
      float cm = fmaxf(s0[r], s1[r]);
      for (int off = 1; off < 16; off <<= 1)
        cm = fmaxf(cm, __shfl_xor(cm, off, 32));
      float nm = fmaxf(mx[r], cm);
      float scale = __expf(mx[r] - nm);
      float p0 = __expf(s0[r] - nm);
      float p1 = __expf(s1[r] - nm);
      float rs = p0 + p1;
      for (int off = 1; off < 16; off <<= 1)
        rs += __shfl_xor(rs, off, 32);
      ls[r] = ls[r] * scale + rs;
      mx[r] = nm;
      o0[r] *= scale; o1[r] *= scale;
      int m = r + khi8;
      pbuf[wave][m][n]      = (_Float16)p0;
      pbuf[wave][m][16 + n] = (_Float16)p1;
    }
    __syncthreads();                      // pbuf ready

    // P as A-fragment (16 queries x 32 keys)
    v16h pa;
#pragma unroll
    for (int i = 0; i < 8; ++i) {
      int k = 2 * i + ((i >= 4) ? 8 : 0) + khi8;
      pa[2 * i]     = pbuf[wave][n][k];
      pa[2 * i + 1] = pbuf[wave][n][k + 1];
    }
    // V B-fragments (k = key, col = dh): key pairs contiguous in kvpan row
    v16h vb0, vb1;
#pragma unroll
    for (int i = 0; i < 8; ++i) {
      int kkey = 2 * i + ((i >= 4) ? 8 : 0) + khi8;
      vb0[2 * i]     = (_Float16)kvpan[32 + n][kkey];
      vb0[2 * i + 1] = (_Float16)kvpan[32 + n][kkey + 1];
      vb1[2 * i]     = (_Float16)kvpan[48 + n][kkey];
      vb1[2 * i + 1] = (_Float16)kvpan[48 + n][kkey + 1];
    }
    o0 = wmma16x16x32(pa, vb0, o0);
    o1 = wmma16x16x32(pa, vb1, o1);
    __syncthreads();                      // reads done; safe to restage
  }

  float* cb = comb + (size_t)b * CIN * HW + (size_t)(HF + h * 32) * HW;
#pragma unroll
  for (int r = 0; r < 8; ++r) {
    float inv = 1.0f / ls[r];
    int q = q0 + r + khi8;
    cb[(size_t)n * HW + q]        = o0[r] * inv;
    cb[(size_t)(16 + n) * HW + q] = o1[r] * inv;
  }
}

// ---------------------------------------------------------------------------
// K5: global-average pool + squeeze-excite gate.
// ---------------------------------------------------------------------------
__global__ __launch_bounds__(256) void k_chan_attn(
    const float* __restrict__ comb, const float* __restrict__ w1,
    const float* __restrict__ w2, float* __restrict__ ca) {
  __shared__ float pooled[CIN];
  __shared__ float hidden[16];
  const int b  = blockIdx.x;
  const int ch = threadIdx.x;
  const float* p = comb + (size_t)b * CIN * HW + (size_t)ch * HW;
  float s = 0.0f;
  for (int i = 0; i < HW; ++i) s += p[i];
  pooled[ch] = s * (1.0f / HW);
  __syncthreads();
  if (ch < 16) {
    float hsum = 0.0f;
    for (int c = 0; c < CIN; ++c) hsum += w1[ch * CIN + c] * pooled[c];
    hidden[ch] = fmaxf(hsum, 0.0f);
  }
  __syncthreads();
  float o = 0.0f;
#pragma unroll
  for (int r = 0; r < 16; ++r) o += w2[ch * 16 + r] * hidden[r];
  ca[b * CIN + ch] = 1.0f / (1.0f + __expf(-o));
}

// ---------------------------------------------------------------------------
// K6: output projection (gate folded into B panel) + residual.
// ---------------------------------------------------------------------------
__global__ __launch_bounds__(256) void k_proj_out(
    const float* __restrict__ x, const float* __restrict__ comb,
    const float* __restrict__ ca, const _Float16* __restrict__ pwz,
    const float* __restrict__ pb, float* __restrict__ out) {
  __shared__ _Float16 pan[16][258];
  const int hw0  = blockIdx.x * 16;
  const int mblk = blockIdx.y;
  const int b    = blockIdx.z;
  const float* cbase = comb + (size_t)b * CIN * HW;
  const float* cav   = ca + b * CIN;
  {
    int n  = threadIdx.x & 15;
    int kb = (threadIdx.x >> 4) * 16;
#pragma unroll
    for (int i = 0; i < 16; ++i) {
      int c = kb + i;
      pan[n][c] = (_Float16)(cbase[(size_t)c * HW + hw0 + n] * cav[c]);
    }
  }
  __syncthreads();

  const int wave = threadIdx.x >> 5;
  const int lane = threadIdx.x & 31;
  const int m0   = mblk * 128 + wave * 16;
  const int tile = m0 >> 4;
  __builtin_prefetch(pwz + (size_t)tile * 8 * 32 * 16, 0, 3);

  v8f acc = {};
#pragma unroll
  for (int ks = 0; ks < 8; ++ks) {
    v16h a  = load_a_swz(pwz, tile, 8, ks, lane);
    v16h bb = load_b_lds(&pan[0][0], 258, ks * 32, lane);
    acc = wmma16x16x32(a, bb, acc);
  }
  const float* xb = x + (size_t)b * CIN * HW;
  float* ob = out + (size_t)b * CIN * HW;
  const int n  = lane & 15;
  const int mh = (lane & 16) ? 8 : 0;
#pragma unroll
  for (int r = 0; r < 8; ++r) {
    int m = m0 + r + mh;
    size_t idx = (size_t)m * HW + hw0 + n;
    ob[idx] = acc[r] + pb[m] + xb[idx];
  }
}

// ---------------------------------------------------------------------------
extern "C" void kernel_launch(void* const* d_in, const int* in_sizes, int n_in,
                              void* d_out, int out_size, void* d_ws, size_t ws_size,
                              hipStream_t stream) {
  const float* x     = (const float*)d_in[0];
  const float* qkv_w = (const float*)d_in[1];
  const float* qkv_b = (const float*)d_in[2];
  const float* dw_w  = (const float*)d_in[3];
  const float* dw_b  = (const float*)d_in[4];
  const float* aw1   = (const float*)d_in[5];
  const float* ab1   = (const float*)d_in[6];
  const float* aw2   = (const float*)d_in[7];
  const float* ab2   = (const float*)d_in[8];
  const float* gq_w  = (const float*)d_in[9];
  const float* gq_b  = (const float*)d_in[10];
  const float* gkv_w = (const float*)d_in[11];
  const float* gkv_b = (const float*)d_in[12];
  const float* ca_w1 = (const float*)d_in[13];
  const float* ca_w2 = (const float*)d_in[14];
  const float* pw    = (const float*)d_in[15];
  const float* pb    = (const float*)d_in[16];
  float* out = (float*)d_out;

  char* ws = (char*)d_ws;
  const size_t S_qkv = (size_t)NB * QKV_CH * HW * sizeof(float);
  float* qkv_raw = (float*)ws;  ws += S_qkv;
  float* qkv_dw  = (float*)ws;  ws += S_qkv;
  float* g_buf   = (float*)ws;  ws += S_qkv;
  float* comb    = (float*)ws;  ws += (size_t)NB * CIN * HW * sizeof(float);
  float* ca      = (float*)ws;  ws += (size_t)NB * CIN * sizeof(float);
  _Float16* w_in_z = (_Float16*)ws;  ws += (size_t)48 * 8 * 32 * 16 * 2;
  _Float16* w1z    = (_Float16*)ws;  ws += (size_t)8 * 4 * 32 * 16 * 2;
  _Float16* w2z    = (_Float16*)ws;  ws += (size_t)8 * 4 * 32 * 16 * 2;
  _Float16* pwz    = (_Float16*)ws;

  // Weight pre-swizzle (f32 -> f16 A-fragment layout)
  k_prep_fused_w<<<dim3(48, 8), 32, 0, stream>>>(qkv_w, gq_w, gkv_w, w_in_z);
  k_prep_w<<<dim3(8, 4), 32, 0, stream>>>(aw1, HF, 4, w1z);
  k_prep_w<<<dim3(8, 4), 32, 0, stream>>>(aw2, HF, 4, w2z);
  k_prep_w<<<dim3(16, 8), 32, 0, stream>>>(pw, CIN, 8, pwz);

  k_fused_in_gemm<<<dim3(HW / 16, 6, NB), 256, 0, stream>>>(
      x, w_in_z, qkv_b, gq_b, gkv_b, qkv_raw, g_buf);
  k_dwconv<<<dim3((NB * QKV_CH * HW) / 256), 256, 0, stream>>>(
      qkv_raw, dw_w, dw_b, qkv_dw);
  k_highfreq<<<dim3(HW / 16, NB), 256, 0, stream>>>(
      qkv_dw, w1z, ab1, w2z, ab2, comb);
  k_lowfreq_attn<<<dim3(8, 4, NB), 256, 0, stream>>>(g_buf, comb);
  k_chan_attn<<<dim3(NB), 256, 0, stream>>>(comb, ca_w1, ca_w2, ca);
  k_proj_out<<<dim3(HW / 16, 2, NB), 256, 0, stream>>>(
      x, comb, ca, pwz, pb, out);
}
